// VectorQuantizer_77309412010
// MI455X (gfx1250) — matrix operand order repaired and checked by reading, not compiled, not run on hardware
//
#include <hip/hip_runtime.h>
#include <hip/hip_bf16.h>

typedef __attribute__((ext_vector_type(16))) _Float16 v16h;
typedef __attribute__((ext_vector_type(8)))  _Float16 v8h;
typedef __attribute__((ext_vector_type(8)))  float    v8f;
typedef __attribute__((ext_vector_type(4)))  unsigned int u32x4;
typedef __attribute__((ext_vector_type(8)))  int          i32x8;
typedef __attribute__((ext_vector_type(4)))  int          i32x4;

#define NUM_EMB 1024
#define EMB_DIM 256
#define BB 16
#define CC 256
#define TT 4096
#define NN (BB * TT)              // 65536 vectors
#define NTOT ((size_t)BB * CC * TT)

// LDS staging geometry: chunks of 64 codes, rows padded to 528B (132 dwords)
// so the 16-lane B-fragment DS reads land on distinct banks (stride 4 banks).
#define CHUNK_CODES 64
#define ROW_HALFS   264            // 256 data + 8 pad halves = 528 bytes
#define NCHUNK      (NUM_EMB / CHUNK_CODES)

// ---------------- Kernel 1: codebook -> f16 + norms ----------------
__global__ void vq_prep_codebook(const float* __restrict__ cb,
                                 _Float16* __restrict__ cbh,
                                 float* __restrict__ cbn) {
    int k = blockIdx.x;           // 1024 blocks
    int c = threadIdx.x;          // 256 threads
    float v = cb[(size_t)k * EMB_DIM + c];
    cbh[(size_t)k * EMB_DIM + c] = (_Float16)v;
    float s = v * v;
    #pragma unroll
    for (int off = 16; off; off >>= 1) s += __shfl_xor(s, off, 32);
    __shared__ float wsum[8];
    if ((threadIdx.x & 31) == 0) wsum[threadIdx.x >> 5] = s;
    __syncthreads();
    if (threadIdx.x == 0) {
        float t = 0.f;
        #pragma unroll
        for (int i = 0; i < 8; ++i) t += wsum[i];
        cbn[k] = t;
    }
}

// ---------------- Kernel 2: [B,C,T] f32 -> xT [B*T, C] f16 ----------------
__global__ void vq_transpose_f16(const float* __restrict__ in,
                                 _Float16* __restrict__ xT) {
    __shared__ float tile[32][33];
    int b = blockIdx.z;
    int c0 = blockIdx.y * 32;
    int t0 = blockIdx.x * 32;
    int tx = threadIdx.x, ty = threadIdx.y;   // (32, 8)
    const float* src = in + (size_t)b * CC * TT;
    #pragma unroll
    for (int i = 0; i < 32; i += 8)
        tile[ty + i][tx] = src[(size_t)(c0 + ty + i) * TT + t0 + tx];
    __syncthreads();
    _Float16* dst = xT + ((size_t)b * TT + t0) * EMB_DIM + c0;
    #pragma unroll
    for (int i = 0; i < 32; i += 8)
        dst[(size_t)(ty + i) * EMB_DIM + tx] = (_Float16)tile[tx][ty + i];
}

// ---------------- TDM: DMA one codebook chunk (64 rows x 512B) into LDS ----
// D# per CDNA5 ISA ch.8: group0 = {flags, lds_addr, global_addr lo/hi|type},
// group1 = {mask/data_size/pad, dims/strides}. data_size=4B. Padding: after
// every 128 dwords (one 512B row) insert 4 dwords -> LDS row stride 528B.
__device__ __forceinline__ void tdm_load_chunk(const _Float16* gsrc,
                                               unsigned int lds_addr) {
    unsigned long long ga = (unsigned long long)(uintptr_t)gsrc;
    u32x4 g0;
    g0[0] = 1u;                                        // count=1, user D#
    g0[1] = lds_addr;                                  // LDS dest (bytes)
    g0[2] = (unsigned int)ga;                          // global_addr[31:0]
    g0[3] = (unsigned int)((ga >> 32) & 0x1FFFFFFu)    // global_addr[56:32]
          | (2u << 30);                                // type = 2 (image)
    i32x8 g1;
    g1[0] = (2 << 16)            // data_size = 4 bytes
          | (1 << 20)            // pad_enable
          | (6 << 22)            // pad_interval: 2^6*8B = 512B (128 dwords)
          | (3 << 25);           // pad_amount: 4 dwords (16B)
    g1[1] = (int)(128u << 16);   // tensor_dim0[15:0]=128 dwords (bits 63:48)
    g1[2] = (int)(64u << 16);    // tensor_dim0 hi=0 | tensor_dim1[15:0]=64
    g1[3] = (int)(128u << 16);   // tensor_dim1 hi=0 | tile_dim0=128 dwords
    g1[4] = 64;                  // tile_dim1=64 rows | tile_dim2=0
    g1[5] = 128;                 // tensor_dim0_stride = 128 dwords
    g1[6] = 0;                   // stride hi | tensor_dim1_stride lo
    g1[7] = 0;
    i32x4 z4 = {0, 0, 0, 0};
#if defined(__clang_major__) && (__clang_major__ >= 23)
    i32x8 z8 = {0, 0, 0, 0, 0, 0, 0, 0};
    __builtin_amdgcn_tensor_load_to_lds(g0, g1, z4, z4, z8, 0);
#else
    __builtin_amdgcn_tensor_load_to_lds(g0, g1, z4, z4, 0);
#endif
}

// ---------------- Kernel 3: WMMA distance + argmin, TDM-staged B ----------
// 4 waves/block; each wave owns TWO 16-row M-tiles (A = 32x256 f16 in 128
// VGPRs, two independent accumulator chains per B fragment). Codebook is
// staged chunk-by-chunk (64 codes = 32KB -> padded 33KB) in double-buffered
// LDS by the Tensor Data Mover, overlapped with compute via TENSORcnt.
// __launch_bounds__(128, 1): allow a large VGPR allocation so the A panel
// stays register-resident (round-2 build spilled it to scratch).
#define WAVES_PER_BLK 4
#define ROWS_PER_WAVE 32
__global__ void __launch_bounds__(128, 1)
vq_argmin_wmma(const _Float16* __restrict__ xT,
               const _Float16* __restrict__ cbh,
               const float* __restrict__ cbn,
               int* __restrict__ idxi,
               float* __restrict__ idxf) {
    __shared__ _Float16 sbuf[2][CHUNK_CODES * ROW_HALFS];   // 2 x 33,792 B

    const int lane = threadIdx.x & 31;
    const int w    = threadIdx.x >> 5;
    const int m0   = (blockIdx.x * WAVES_PER_BLK + w) * ROWS_PER_WAVE;
    const int half = lane >> 4;       // 0: lanes 0-15, 1: lanes 16-31
    const int lm   = lane & 15;

    const unsigned int lds_base = (unsigned int)(uintptr_t)&sbuf[0][0];
    const unsigned int buf_bytes = CHUNK_CODES * ROW_HALFS * 2;

    // Prime the pipeline first: wave 0 issues TDM for chunk 0 so the DMA
    // overlaps the A-panel global loads below.
    if (threadIdx.x < 32) tdm_load_chunk(cbh, lds_base);

    // A panels (ISA 16-bit A layout): lanes 0-15 carry M=lm with
    // K {j*32+0..7, j*32+16..23}; lanes 16-31 carry K {+8, +24}.
    v16h a[2][8];
    #pragma unroll
    for (int t = 0; t < 2; ++t) {
        const _Float16* abase =
            xT + (size_t)(m0 + t * 16 + lm) * EMB_DIM + half * 8;
        #pragma unroll
        for (int j = 0; j < 8; ++j) {
            v8h lo = *(const v8h*)(abase + j * 32);
            v8h hi = *(const v8h*)(abase + j * 32 + 16);
            #pragma unroll
            for (int i = 0; i < 8; ++i) { a[t][j][i] = lo[i]; a[t][j][i + 8] = hi[i]; }
        }
    }

    float minv[2][8]; int mini[2][8];
    #pragma unroll
    for (int t = 0; t < 2; ++t)
        #pragma unroll
        for (int r = 0; r < 8; ++r) { minv[t][r] = 3.4e38f; mini[t][r] = 0; }

    for (int ch = 0; ch < NCHUNK; ++ch) {
        if (threadIdx.x < 32) {
            __builtin_amdgcn_s_wait_tensorcnt(0);    // chunk ch landed in LDS
        }
        __syncthreads();
        if (threadIdx.x < 32 && (ch + 1) < NCHUNK) { // prefetch chunk ch+1
            tdm_load_chunk(cbh + (size_t)(ch + 1) * CHUNK_CODES * EMB_DIM,
                           lds_base + ((unsigned)(ch + 1) & 1u) * buf_bytes);
        }
        const _Float16* buf = &sbuf[ch & 1][0];
        const int k0 = ch * CHUNK_CODES;

        #pragma unroll
        for (int kt = 0; kt < CHUNK_CODES / 16; ++kt) {
            // B tile (32x16) from LDS: lane lm = column (code), lanes 0-15
            // carry K rows d0+{0..7,16..23}, lanes 16-31 d0+{8..15,24..31}.
            const _Float16* bb = buf + (kt * 16 + lm) * ROW_HALFS + half * 8;
            v8f c0 = {}, c1 = {};
            #pragma unroll
            for (int j = 0; j < 8; ++j) {
                v8h lo = *(const v8h*)(bb + j * 32);
                v8h hi = *(const v8h*)(bb + j * 32 + 16);
                v16h bfrag;
                #pragma unroll
                for (int i = 0; i < 8; ++i) { bfrag[i] = lo[i]; bfrag[i + 8] = hi[i]; }
                c0 = __builtin_amdgcn_wmma_f32_16x16x32_f16(
                         false, a[0][j], false, bfrag, (short)0, c0, false, false);
                c1 = __builtin_amdgcn_wmma_f32_16x16x32_f16(
                         false, a[1][j], false, bfrag, (short)0, c1, false, false);
            }
            float nrm = cbn[k0 + kt * 16 + lm];
            int   kk  = k0 + kt * 16 + lm;
            #pragma unroll
            for (int r = 0; r < 8; ++r) {
                float d0 = __builtin_fmaf(-2.0f, c0[r], nrm);
                float d1 = __builtin_fmaf(-2.0f, c1[r], nrm);
                if (d0 < minv[0][r]) { minv[0][r] = d0; mini[0][r] = kk; }
                if (d1 < minv[1][r]) { minv[1][r] = d1; mini[1][r] = kk; }
            }
        }
        __syncthreads();   // all waves done with buf before TDM reuses it
    }

    // Reduce (min, argmin) across the 16 lanes sharing each row group.
    #pragma unroll
    for (int t = 0; t < 2; ++t) {
        #pragma unroll
        for (int r = 0; r < 8; ++r) {
            #pragma unroll
            for (int off = 8; off; off >>= 1) {
                float ov = __shfl_xor(minv[t][r], off, 32);
                int   oi = __shfl_xor(mini[t][r], off, 32);
                if (ov < minv[t][r] || (ov == minv[t][r] && oi < mini[t][r])) {
                    minv[t][r] = ov; mini[t][r] = oi;
                }
            }
        }
        if (lm == 0) {
            #pragma unroll
            for (int r = 0; r < 8; ++r) {
                int m = m0 + t * 16 + half * 8 + r;
                idxi[m] = mini[t][r];
                idxf[m] = (float)mini[t][r];
            }
        }
    }
}

// ---------------- Kernel 4: gather codes -> [B,C,T] + loss partials -------
__global__ void vq_gather_loss(const float* __restrict__ in,
                               const float* __restrict__ cb,
                               const int* __restrict__ idxi,
                               float* __restrict__ outq,
                               float* __restrict__ partials) {
    int b = blockIdx.z;
    int c0 = blockIdx.y * 32;
    int t0 = blockIdx.x * 32;
    int tx = threadIdx.x, ty = threadIdx.y;   // (32, 8)
    __shared__ int sidx[32];
    if (ty == 0) sidx[tx] = idxi[b * TT + t0 + tx];
    __syncthreads();
    const float* src = in + (size_t)b * CC * TT;
    float* dst = outq + (size_t)b * CC * TT;
    int k = sidx[tx];
    float acc = 0.f;
    #pragma unroll
    for (int i = 0; i < 32; i += 8) {
        int c = c0 + ty + i;
        float q = cb[(size_t)k * EMB_DIM + c];
        size_t off = (size_t)c * TT + t0 + tx;
        float x = src[off];
        dst[off] = q;
        float d = q - x;
        acc += d * d;
    }
    #pragma unroll
    for (int off = 16; off; off >>= 1) acc += __shfl_xor(acc, off, 32);
    __shared__ float wsum[8];
    if (tx == 0) wsum[ty] = acc;
    __syncthreads();
    if (tx == 0 && ty == 0) {
        float t = 0.f;
        #pragma unroll
        for (int i = 0; i < 8; ++i) t += wsum[i];
        int bid = (blockIdx.z * gridDim.y + blockIdx.y) * gridDim.x + blockIdx.x;
        partials[bid] = t;
    }
}

// ---------------- Kernel 5: reduce partials -> vq_loss ----------------
__global__ void vq_reduce_loss(const float* __restrict__ partials,
                               float* __restrict__ loss, int n) {
    float s = 0.f;
    for (int i = threadIdx.x; i < n; i += 256) s += partials[i];
    #pragma unroll
    for (int off = 16; off; off >>= 1) s += __shfl_xor(s, off, 32);
    __shared__ float wsum[8];
    if ((threadIdx.x & 31) == 0) wsum[threadIdx.x >> 5] = s;
    __syncthreads();
    if (threadIdx.x == 0) {
        float t = 0.f;
        #pragma unroll
        for (int i = 0; i < 8; ++i) t += wsum[i];
        // vq_loss = e_latent + 0.25*q_latent = 1.25 * mean(diff^2)
        *loss = t * (1.25f / (float)NTOT);
    }
}

extern "C" void kernel_launch(void* const* d_in, const int* in_sizes, int n_in,
                              void* d_out, int out_size, void* d_ws, size_t ws_size,
                              hipStream_t stream) {
    const float* inp = (const float*)d_in[0];   // [B, C, T] f32
    const float* cb  = (const float*)d_in[1];   // [K, D]   f32

    // Workspace layout (256B-aligned offsets)
    char* ws = (char*)d_ws;
    _Float16* xT   = (_Float16*)(ws + 0);            // 65536*256*2 = 33,554,432
    _Float16* cbh  = (_Float16*)(ws + 33554432);     // 1024*256*2  =    524,288
    float*    cbn  = (float*)   (ws + 34078720);     // 1024*4      =      4,096
    int*      idxi = (int*)     (ws + 34082816);     // 65536*4     =    262,144
    float*    part = (float*)   (ws + 34344960);     // 16384*4     =     65,536

    // Output layout: quantized_st [B*C*T] | vq_loss [1] | indices [B*T] (as f32)
    float* outq = (float*)d_out;
    float* loss = outq + NTOT;
    float* idxf = loss + 1;

    vq_prep_codebook<<<NUM_EMB, EMB_DIM, 0, stream>>>(cb, cbh, cbn);
    vq_transpose_f16<<<dim3(TT / 32, CC / 32, BB), dim3(32, 8), 0, stream>>>(inp, xT);
    vq_argmin_wmma<<<NN / (WAVES_PER_BLK * ROWS_PER_WAVE), 32 * WAVES_PER_BLK, 0,
                     stream>>>(xT, cbh, cbn, idxi, idxf);
    vq_gather_loss<<<dim3(TT / 32, CC / 32, BB), dim3(32, 8), 0, stream>>>(
        inp, cb, idxi, outq, part);
    vq_reduce_loss<<<1, 256, 0, stream>>>(part, loss, (TT / 32) * (CC / 32) * BB);
}
